// AffinityPredictor_13065290514621
// MI455X (gfx1250) — matrix operand order repaired and checked by reading, not compiled, not run on hardware
//
#include <hip/hip_runtime.h>
#include <math.h>

// ---------------------------------------------------------------------------
// CDNA5 (gfx1250) implementation of the AffinityPredictor forward pass.
// Dense GEMMs -> v_wmma_f32_16x16x32_f16, two 16x16 C tiles per wave,
// B tiles staged cache->LDS with GLOBAL_LOAD_ASYNC_TO_LDS_B128 (ASYNCcnt),
// global_prefetch_b8 on the next A k-slab.
// Irregular graph segment ops -> VALU + global atomics.
// ---------------------------------------------------------------------------

typedef __attribute__((ext_vector_type(16))) _Float16 v16h;
typedef __attribute__((ext_vector_type(8)))  float    v8f;

#define CONSTS \
  constexpr int NB = 32; constexpr int NN = 10000; constexpr int NE = 160000; \
  constexpr int HID = 128; constexpr int OC = 256; constexpr int HOC = 512;   \
  constexpr int EDIM = 32; constexpr int SEQD = 1280;                         \
  constexpr int CH = 20000; constexpr int NCH = 8;

// monotone float<->uint key for lock-free atomic max on floats
__device__ __forceinline__ unsigned f2key(float f) {
  unsigned u = __float_as_uint(f);
  return (u & 0x80000000u) ? ~u : (u | 0x80000000u);
}
__device__ __forceinline__ float key2f(unsigned k) {
  unsigned u = (k & 0x80000000u) ? (k & 0x7FFFFFFFu) : ~k;
  return __uint_as_float(u);
}

// ---------------------------------------------------------------------------
// Generic WMMA GEMM: C[M,N] = act(A[M,K] @ W[K,N] + bias)
// act: 0 = none, 1 = relu, 2 = elu. N must be a multiple of 32.
// Block = 128 threads = 4 waves; wave w owns rows m0..m0+15 with
// m0=(blockIdx.x*4+w)*16 and the 32-column strip n0 = blockIdx.y*32
// (two 16x16 WMMA accumulators sharing one A fragment).
// The 32(K)x32(N) B tile is staged into LDS via async b128 loads.
// ---------------------------------------------------------------------------
__global__ __launch_bounds__(128) void gemm_wmma_f16(
    const float* __restrict__ A, const float* __restrict__ W,
    const float* __restrict__ bias, float* __restrict__ C,
    int M, int N, int K, int act)
{
  __shared__ float Bs[32 * 32];          // [kk][nn], 4 KB, LDS offset 0
  const int wave  = threadIdx.x >> 5;
  const int lane  = threadIdx.x & 31;
  const int half  = lane >> 4;           // 0: lanes 0-15, 1: lanes 16-31
  const int idx16 = lane & 15;
  const int m0 = (blockIdx.x * 4 + wave) * 16;
  const int n0 = blockIdx.y * 32;
  const bool active = (m0 < M);
  const int arow = m0 + idx16;

  v8f acc0 = {}, acc1 = {};
  for (int kb = 0; kb < K; kb += 32) {
    // ---- stage B tile: 256 float4 transfers, 2 per thread, async into LDS
#pragma unroll
    for (int r = 0; r < 2; ++r) {
      int j   = threadIdx.x + r * 128;   // float4 index 0..255
      int kk  = j >> 3;                  // B-tile row (k)
      int nn4 = (j & 7) * 4;             // starting col of this float4
      int kg  = kb + kk;
      if (kg < K) {
        unsigned ldsoff = (unsigned)(j * 16);          // byte offset in Bs
        const float* gp = W + (size_t)kg * N + (n0 + nn4);
        asm volatile("global_load_async_to_lds_b128 %0, %1, off"
                     :: "v"(ldsoff), "v"(gp) : "memory");
      } else {                            // zero-pad rows past K
        Bs[kk * 32 + nn4 + 0] = 0.0f;
        Bs[kk * 32 + nn4 + 1] = 0.0f;
        Bs[kk * 32 + nn4 + 2] = 0.0f;
        Bs[kk * 32 + nn4 + 3] = 0.0f;
      }
    }
    asm volatile("s_wait_asynccnt 0" ::: "memory");     // our async writes landed
    __syncthreads();                                    // all waves' tiles visible

    if (active) {
      if (kb + 32 < K)  // gfx1250 global_prefetch_b8 on next A slab
        __builtin_prefetch(&A[(size_t)arow * K + kb + 32], 0, 1);
      v16h a, b0, b1;
#pragma unroll
      for (int j = 0; j < 16; ++j) {
        // 16-bit A 16x32 layout: lanes<16 -> K 0..7,16..23 ; lanes>=16 -> K 8..15,24..31
        int ka = kb + half * 8 + (j & 7) + ((j >> 3) << 4);
        float av = (ka < K) ? A[(size_t)arow * K + ka] : 0.0f;
        a[j] = (_Float16)av;
        // B 32x16 layout: lanes<16 -> K 0..7,16..23 ; lanes>=16 -> K 8..15,24..31 ; col = lane%16
        int krow = half * 8 + (j & 7) + ((j >> 3) << 4);
        b0[j] = (_Float16)Bs[krow * 32 + idx16];
        b1[j] = (_Float16)Bs[krow * 32 + 16 + idx16];
      }
      acc0 = __builtin_amdgcn_wmma_f32_16x16x32_f16(false, a, false, b0,
                                                    (short)0, acc0, false, false);
      acc1 = __builtin_amdgcn_wmma_f32_16x16x32_f16(false, a, false, b1,
                                                    (short)0, acc1, false, false);
    }
    __syncthreads();
  }
  if (active) {
    const int na = n0 + idx16;
    const int nbcol = n0 + 16 + idx16;
    const float bv0 = bias ? bias[na] : 0.0f;
    const float bv1 = bias ? bias[nbcol] : 0.0f;
#pragma unroll
    for (int r = 0; r < 8; ++r) {        // D layout: M = r + 8*half, N = lane%16
      int m = m0 + r + 8 * half;
      float v0 = acc0[r] + bv0;
      float v1 = acc1[r] + bv1;
      if (act == 1)      { v0 = fmaxf(v0, 0.0f); v1 = fmaxf(v1, 0.0f); }
      else if (act == 2) {
        v0 = (v0 > 0.0f) ? v0 : (__expf(v0) - 1.0f);
        v1 = (v1 > 0.0f) ? v1 : (__expf(v1) - 1.0f);
      }
      if (m < M) {
        C[(size_t)m * N + na]    = v0;
        C[(size_t)m * N + nbcol] = v1;
      }
    }
  }
}

// ---------------------------------------------------------------------------
// TransformerConv edge attention: alpha[e,h] = q[dst,h,:]·(k[src,h,:]+ee2[e,h,:]) / 16
// plus atomic segment max over dst.
// ---------------------------------------------------------------------------
__global__ void edge_alpha_k(const float* __restrict__ q, const float* __restrict__ k0,
                             const float* __restrict__ ee2c, const int* __restrict__ ei,
                             int e0, int ecount, int etot,
                             float* __restrict__ alpha, unsigned* __restrict__ mkey)
{
  CONSTS; (void)NB; (void)NN; (void)NE; (void)HID; (void)EDIM; (void)SEQD; (void)CH; (void)NCH;
  int t = blockIdx.x * blockDim.x + threadIdx.x;
  if (t >= ecount * 2) return;
  int el = t >> 1, h = t & 1;
  int e = e0 + el;
  int src = ei[e], dst = ei[etot + e];
  const float* qp = q    + (size_t)dst * HOC + h * OC;
  const float* kp = k0   + (size_t)src * HOC + h * OC;
  const float* ep = ee2c + (size_t)el * HOC + h * OC;
  float d = 0.0f;
  for (int c = 0; c < OC; ++c) d += qp[c] * (kp[c] + ep[c]);
  d *= 0.0625f;                       // 1/sqrt(256)
  alpha[(size_t)e * 2 + h] = d;
  atomicMax(&mkey[dst * 2 + h], f2key(d));
}

__global__ void edge_expsum_k(const int* __restrict__ ei, int etot,
                              float* __restrict__ alpha, const unsigned* __restrict__ mkey,
                              float* __restrict__ den)
{
  int t = blockIdx.x * blockDim.x + threadIdx.x;
  if (t >= etot * 2) return;
  int e = t >> 1, h = t & 1;
  int dst = ei[etot + e];
  float ex = __expf(alpha[(size_t)e * 2 + h] - key2f(mkey[dst * 2 + h]));
  alpha[(size_t)e * 2 + h] = ex;
  atomicAdd(&den[dst * 2 + h], ex);
}

__global__ void edge_scatter_k(const float* __restrict__ v0, const float* __restrict__ ee2c,
                               const int* __restrict__ ei, int e0, int ecount, int etot,
                               const float* __restrict__ alpha, const float* __restrict__ den,
                               float* __restrict__ attn)
{
  CONSTS; (void)NB; (void)NN; (void)NE; (void)HID; (void)EDIM; (void)SEQD; (void)CH; (void)NCH;
  int t = blockIdx.x * blockDim.x + threadIdx.x;
  if (t >= ecount * 2) return;
  int el = t >> 1, h = t & 1;
  int e = e0 + el;
  int src = ei[e], dst = ei[etot + e];
  float a = alpha[(size_t)e * 2 + h] / (den[dst * 2 + h] + 1e-16f);
  const float* vp = v0   + (size_t)src * HOC + h * OC;
  const float* ep = ee2c + (size_t)el * HOC + h * OC;
  float* op = attn + (size_t)dst * HOC + h * OC;
  for (int c = 0; c < OC; ++c) atomicAdd(&op[c], a * (vp[c] + ep[c]));
}

// h = elu(mean_heads(attn) + skip)
__global__ void elu_mean_skip_k(const float* __restrict__ attn, const float* __restrict__ skip,
                                float* __restrict__ hbuf, int n)
{
  CONSTS; (void)NB; (void)NN; (void)NE; (void)HID; (void)EDIM; (void)SEQD; (void)CH; (void)NCH;
  int t = blockIdx.x * blockDim.x + threadIdx.x;
  if (t >= n * OC) return;
  int node = t >> 8, c = t & 255;
  float v = 0.5f * (attn[(size_t)node * HOC + c] + attn[(size_t)node * HOC + OC + c]) + skip[t];
  hbuf[t] = (v > 0.0f) ? v : (__expf(v) - 1.0f);
}

// ---------------------------------------------------------------------------
// Sequence-guided attention pooling
// ---------------------------------------------------------------------------
__global__ void pool_score_k(const float* __restrict__ hq, const float* __restrict__ ctxk,
                             const float* __restrict__ pv, const int* __restrict__ batch,
                             float* __restrict__ sarr, unsigned* __restrict__ smaxkey, int n)
{
  CONSTS; (void)NB; (void)NN; (void)NE; (void)OC; (void)HOC; (void)EDIM; (void)SEQD; (void)CH; (void)NCH;
  int i = blockIdx.x * blockDim.x + threadIdx.x;
  if (i >= n) return;
  int b = batch[i];
  const float* hp = hq + (size_t)i * HID;
  const float* cp = ctxk + (size_t)b * HID;
  float s = 0.0f;
  for (int d = 0; d < HID; ++d) s += tanhf(hp[d] + cp[d]) * pv[d];
  sarr[i] = s;
  atomicMax(&smaxkey[b], f2key(s));
}

__global__ void pool_norm_k(const int* __restrict__ batch, const float* __restrict__ sarr,
                            const unsigned* __restrict__ smaxkey, float* __restrict__ sden,
                            float* __restrict__ exn, int n)
{
  int i = blockIdx.x * blockDim.x + threadIdx.x;
  if (i >= n) return;
  int b = batch[i];
  float ex = __expf(sarr[i] - key2f(smaxkey[b]));
  exn[i] = ex;
  atomicAdd(&sden[b], ex);
}

__global__ void pool_scatter_k(const int* __restrict__ batch, const float* __restrict__ exn,
                               const float* __restrict__ sden, const float* __restrict__ nodeh,
                               float* __restrict__ pooled, int n)
{
  CONSTS; (void)NB; (void)NN; (void)NE; (void)OC; (void)HOC; (void)EDIM; (void)SEQD; (void)CH; (void)NCH;
  int i = blockIdx.x * blockDim.x + threadIdx.x;
  if (i >= n) return;
  int b = batch[i];
  float a = exn[i] / (sden[b] + 1e-16f);
  const float* hp = nodeh + (size_t)i * HID;
  float* pp = pooled + (size_t)b * HID;
  for (int d = 0; d < HID; ++d) atomicAdd(&pp[d], a * hp[d]);
}

// ---------------------------------------------------------------------------
// Final MLP heads: one block (128 threads) per batch element, LDS reductions.
// ---------------------------------------------------------------------------
__device__ float mlp2_eval(const float* in, int din,
                           const float* __restrict__ w1, const float* __restrict__ b1,
                           const float* __restrict__ w2, const float* __restrict__ b2,
                           float* red)
{
  int t = threadIdx.x;                // 128 hidden units, one per thread
  float acc = b1[t];
  for (int i = 0; i < din; ++i) acc += in[i] * w1[i * 128 + t];
  acc = fmaxf(acc, 0.0f);
  red[t] = acc * w2[t];
  __syncthreads();
  for (int s = 64; s > 0; s >>= 1) {
    if (t < s) red[t] += red[t + s];
    __syncthreads();
  }
  float r = red[0] + b2[0];
  __syncthreads();
  return r;
}

__global__ __launch_bounds__(128) void final_head_k(
    const float* __restrict__ poolL, const float* __restrict__ poolH, const float* __restrict__ poolA,
    const float* __restrict__ ctxL,  const float* __restrict__ ctxH,  const float* __restrict__ ctxA,
    const float* __restrict__ ctxWL, const float* __restrict__ ctxWH, const float* __restrict__ ctxWA,
    const float* __restrict__ dg_w1, const float* __restrict__ dg_b1,
    const float* __restrict__ dg_w2, const float* __restrict__ dg_b2,
    const float* __restrict__ ddg_w1, const float* __restrict__ ddg_b1,
    const float* __restrict__ ddg_w2, const float* __restrict__ ddg_b2,
    const float* __restrict__ inter_w1, const float* __restrict__ inter_b1,
    const float* __restrict__ inter_w2, const float* __restrict__ inter_b2,
    float* __restrict__ out)
{
  __shared__ float gf[768], itf[512], dd[768], red[128];
  int t = threadIdx.x, b = blockIdx.x;
  float l2 = poolL[b * 128 + t], h2 = poolH[b * 128 + t], a2 = poolA[b * 128 + t];
  float lc = ctxL[b * 128 + t],  hc = ctxH[b * 128 + t],  ac = ctxA[b * 128 + t];
  gf[t] = l2; gf[128 + t] = lc; gf[256 + t] = h2; gf[384 + t] = hc; gf[512 + t] = a2; gf[640 + t] = ac;
  itf[t] = l2 * a2; itf[128 + t] = h2 * a2; itf[256 + t] = fabsf(l2 - a2); itf[384 + t] = fabsf(h2 - a2);
  dd[t] = 0.0f; dd[128 + t] = 0.0f; dd[256 + t] = 0.0f;           // diff = zeros(3*HID)
  dd[384 + t] = lc - ctxWL[b * 128 + t];
  dd[512 + t] = hc - ctxWH[b * 128 + t];
  dd[640 + t] = ac - ctxWA[b * 128 + t];
  __syncthreads();
  float r = mlp2_eval(gf, 768, dg_w1, dg_b1, dg_w2, dg_b2, red)                // pred_dg
          + mlp2_eval(itf, 512, inter_w1, inter_b1, inter_w2, inter_b2, red)   // inter_bias
          + mlp2_eval(dd, 768, ddg_w1, ddg_b1, ddg_w2, ddg_b2, red);           // pred_ddg
  if (t == 0) out[b] = r;
}

// ---------------------------------------------------------------------------
// Host orchestration
// ---------------------------------------------------------------------------
static void gemm(hipStream_t s, const float* A, const float* W, const float* bias,
                 float* C, int M, int N, int K, int act)
{
  dim3 grid((M + 63) / 64, N / 32);
  gemm_wmma_f16<<<grid, 128, 0, s>>>(A, W, bias, C, M, N, K, act);
}

extern "C" void kernel_launch(void* const* d_in, const int* in_sizes, int n_in,
                              void* d_out, int out_size, void* d_ws, size_t ws_size,
                              hipStream_t stream)
{
  CONSTS;
  (void)in_sizes; (void)n_in; (void)out_size; (void)ws_size;

  const float* gx[3]  = {(const float*)d_in[0], (const float*)d_in[4], (const float*)d_in[8]};
  const int*   gei[3] = {(const int*)d_in[1],   (const int*)d_in[5],   (const int*)d_in[9]};
  const float* gea[3] = {(const float*)d_in[2], (const float*)d_in[6], (const float*)d_in[10]};
  const int*   gbi[3] = {(const int*)d_in[3],   (const int*)d_in[7],   (const int*)d_in[11]};
  const float* seqs[6] = {(const float*)d_in[12], (const float*)d_in[13], (const float*)d_in[14],
                          (const float*)d_in[15], (const float*)d_in[16], (const float*)d_in[17]};
  // params (make_params insertion order)
  const float* ee_w1  = (const float*)d_in[18]; const float* ee_b1  = (const float*)d_in[19];
  const float* ee_w2  = (const float*)d_in[20]; const float* ee_b2  = (const float*)d_in[21];
  const float* c_wq   = (const float*)d_in[22]; const float* c_bq   = (const float*)d_in[23];
  const float* c_wk   = (const float*)d_in[24]; const float* c_bk   = (const float*)d_in[25];
  const float* c_wv   = (const float*)d_in[26]; const float* c_bv   = (const float*)d_in[27];
  const float* c_we   = (const float*)d_in[28];
  const float* c_wsk  = (const float*)d_in[29]; const float* c_bsk  = (const float*)d_in[30];
  const float* p1_w   = (const float*)d_in[31]; const float* p1_b   = (const float*)d_in[32];
  const float* seq_w  = (const float*)d_in[33]; const float* seq_b  = (const float*)d_in[34];
  const float* pool_wq = (const float*)d_in[35];
  const float* pool_wk = (const float*)d_in[36];
  const float* pool_v  = (const float*)d_in[37];
  const float* dg_w1 = (const float*)d_in[38]; const float* dg_b1 = (const float*)d_in[39];
  const float* dg_w2 = (const float*)d_in[40]; const float* dg_b2 = (const float*)d_in[41];
  const float* ddg_w1 = (const float*)d_in[42]; const float* ddg_b1 = (const float*)d_in[43];
  const float* ddg_w2 = (const float*)d_in[44]; const float* ddg_b2 = (const float*)d_in[45];
  // gate MLP (d_in[46..49]) is computed but unused in the reference output
  const float* in_w1 = (const float*)d_in[50]; const float* in_b1 = (const float*)d_in[51];
  const float* in_w2 = (const float*)d_in[52]; const float* in_b2 = (const float*)d_in[53];

  // -------- workspace carve-up (floats) --------
  float* wsf = (float*)d_ws;
  size_t off = 0;
  auto take = [&](size_t n) { float* p = wsf + off; off += n; return p; };
  float* qb    = take((size_t)NN * HOC);
  float* kb0   = take((size_t)NN * HOC);
  float* vb    = take((size_t)NN * HOC);
  float* skipb = take((size_t)NN * OC);
  float* eeh1  = take((size_t)NE * EDIM);
  float* eebuf = take((size_t)NE * EDIM);
  float* alpha = take((size_t)NE * 2);
  unsigned* mkey = (unsigned*)take((size_t)NN * 2);
  float* den   = take((size_t)NN * 2);
  float* attn  = take((size_t)NN * HOC);
  float* hbuf  = take((size_t)NN * OC);
  float* nodeh = take((size_t)NN * HID);
  float* hqb   = take((size_t)NN * HID);
  float* sarr  = take((size_t)NN);
  float* exn   = take((size_t)NN);
  unsigned* smaxkey = (unsigned*)take(NB);
  float* sden  = take(NB);
  float* ee2c  = take((size_t)CH * HOC);
  float* ctx6  = take((size_t)6 * NB * HID);
  float* ctxk  = take((size_t)NB * HID);
  float* pooled3 = take((size_t)3 * NB * HID);

  // 1) sequence encoders (shared weights): ctx6[i] = seqs[i] @ seq_w + seq_b
  for (int i = 0; i < 6; ++i)
    gemm(stream, seqs[i], seq_w, seq_b, ctx6 + (size_t)i * NB * HID, NB, HID, SEQD, 0);

  // 2) per-graph encode + TransformerConv + pooling
  for (int g = 0; g < 3; ++g) {
    const float* x = gx[g]; const int* ei = gei[g];
    const float* ea = gea[g]; const int* bidx = gbi[g];
    const float* ctx = ctx6 + (size_t)g * NB * HID;
    float* pooled = pooled3 + (size_t)g * NB * HID;

    gemm(stream, ea, ee_w1, ee_b1, eeh1, NE, EDIM, 2, 1);        // relu
    gemm(stream, eeh1, ee_w2, ee_b2, eebuf, NE, EDIM, EDIM, 0);
    gemm(stream, x, c_wq, c_bq, qb, NN, HOC, 23, 0);
    gemm(stream, x, c_wk, c_bk, kb0, NN, HOC, 23, 0);
    gemm(stream, x, c_wv, c_bv, vb, NN, HOC, 23, 0);
    gemm(stream, x, c_wsk, c_bsk, skipb, NN, OC, 23, 0);

    hipMemsetAsync(mkey, 0, (size_t)NN * 2 * sizeof(unsigned), stream);  // key 0 == floor
    hipMemsetAsync(den, 0, (size_t)NN * 2 * sizeof(float), stream);
    hipMemsetAsync(attn, 0, (size_t)NN * HOC * sizeof(float), stream);

    // pass 1: chunked edge GEMM + alpha + segment max
    for (int c = 0; c < NCH; ++c) {
      gemm(stream, eebuf + (size_t)c * CH * EDIM, c_we, nullptr, ee2c, CH, HOC, EDIM, 0);
      edge_alpha_k<<<(CH * 2 + 255) / 256, 256, 0, stream>>>(qb, kb0, ee2c, ei, c * CH, CH, NE, alpha, mkey);
    }
    // pass 2: exp + segment sum
    edge_expsum_k<<<(NE * 2 + 255) / 256, 256, 0, stream>>>(ei, NE, alpha, mkey, den);
    // pass 3: chunked edge GEMM again + weighted scatter
    for (int c = 0; c < NCH; ++c) {
      gemm(stream, eebuf + (size_t)c * CH * EDIM, c_we, nullptr, ee2c, CH, HOC, EDIM, 0);
      edge_scatter_k<<<(CH * 2 + 255) / 256, 256, 0, stream>>>(vb, ee2c, ei, c * CH, CH, NE, alpha, den, attn);
    }

    elu_mean_skip_k<<<((size_t)NN * OC + 255) / 256, 256, 0, stream>>>(attn, skipb, hbuf, NN);
    gemm(stream, hbuf, p1_w, p1_b, nodeh, NN, HID, OC, 0);
    gemm(stream, nodeh, pool_wq, nullptr, hqb, NN, HID, HID, 0);
    gemm(stream, ctx, pool_wk, nullptr, ctxk, NB, HID, HID, 0);

    hipMemsetAsync(smaxkey, 0, NB * sizeof(unsigned), stream);
    hipMemsetAsync(sden, 0, NB * sizeof(float), stream);
    hipMemsetAsync(pooled, 0, (size_t)NB * HID * sizeof(float), stream);
    pool_score_k<<<(NN + 255) / 256, 256, 0, stream>>>(hqb, ctxk, pool_v, bidx, sarr, smaxkey, NN);
    pool_norm_k<<<(NN + 255) / 256, 256, 0, stream>>>(bidx, sarr, smaxkey, sden, exn, NN);
    pool_scatter_k<<<(NN + 255) / 256, 256, 0, stream>>>(bidx, exn, sden, nodeh, pooled, NN);
  }

  // 3) heads: pred_dg + inter_bias + pred_ddg
  final_head_k<<<NB, 128, 0, stream>>>(
      pooled3, pooled3 + (size_t)NB * HID, pooled3 + (size_t)2 * NB * HID,
      ctx6, ctx6 + (size_t)NB * HID, ctx6 + (size_t)2 * NB * HID,
      ctx6 + (size_t)3 * NB * HID, ctx6 + (size_t)4 * NB * HID, ctx6 + (size_t)5 * NB * HID,
      dg_w1, dg_b1, dg_w2, dg_b2,
      ddg_w1, ddg_b1, ddg_w2, ddg_b2,
      in_w1, in_b1, in_w2, in_b2,
      (float*)d_out);
}
